// D2FAgg_16604343566507
// MI455X (gfx1250) — matrix operand order repaired and checked by reading, not compiled, not run on hardware
//
#include <hip/hip_runtime.h>
#include <hip/hip_bf16.h>
#include <math.h>

#define BB   4
#define NN   6144
#define CC   256
#define MM   3
#define NLOC 2048   // n = N / M

typedef __bf16 bf16_t;
typedef bf16_t v16bf __attribute__((ext_vector_type(16)));
typedef bf16_t v8bf  __attribute__((ext_vector_type(8)));
typedef bf16_t v4bf  __attribute__((ext_vector_type(4)));
typedef float  v8f   __attribute__((ext_vector_type(8)));

// ---------------------------------------------------------------------------
// f32 -> bf16 conversion (natural layout)
// ---------------------------------------------------------------------------
__global__ __launch_bounds__(256) void f2bf_kernel(
    const float* __restrict__ in, bf16_t* __restrict__ out, int nElem)
{
    int i = blockIdx.x * 256 + threadIdx.x;
    if (i < nElem) out[i] = (bf16_t)in[i];
}

// ---------------------------------------------------------------------------
// feature_node -> bf16 transposed: xT[(b*M+im)*C + c][node] = x[b][im*n+node][c]
// ---------------------------------------------------------------------------
__global__ __launch_bounds__(256) void xT_kernel(
    const float* __restrict__ xf, bf16_t* __restrict__ xT)
{
    int i = blockIdx.x * 256 + threadIdx.x;     // (bi*C + c)*n + node, node fastest
    if (i >= BB * MM * CC * NLOC) return;
    int node = i % NLOC;
    int c    = (i / NLOC) % CC;
    int bi   = i / (NLOC * CC);
    int im   = bi % MM;
    int b    = bi / MM;
    xT[i] = (bf16_t)xf[((size_t)b * NN + (size_t)im * NLOC + node) * CC + c];
}

// ---------------------------------------------------------------------------
// weight -> bf16 transposed: wT[im][c2][c1] = w[im][c1][c2]
// ---------------------------------------------------------------------------
__global__ __launch_bounds__(256) void wT_kernel(
    const float* __restrict__ w, bf16_t* __restrict__ wT)
{
    int i = blockIdx.x * 256 + threadIdx.x;     // (im*C + c2)*C + c1, c1 fastest
    if (i >= MM * CC * CC) return;
    int c1 = i % CC;
    int c2 = (i / CC) % CC;
    int im = i / (CC * CC);
    wT[i] = (bf16_t)w[((size_t)im * CC + c1) * CC + c2];
}

// ---------------------------------------------------------------------------
// Fused kernel: P = invn * (E @ X - diag*x)   with L1 row-norm computed inline.
// Block = 64 rows x 256 cols (full C), so each E element is read exactly ONCE.
// Diagonal handled post-loop via 64 direct loads (no compares in hot loop).
// 8 waves: waveRow (2) x waveCol (4); each wave 32x64 = 2x4 wmma tiles.
// Grid (n/64, 1, B*M).
// ---------------------------------------------------------------------------
__global__ __launch_bounds__(256) void eggemm_kernel(
    const float*  __restrict__ E,  const bf16_t* __restrict__ xT,
    const float*  __restrict__ xf, bf16_t* __restrict__ P)
{
    const int tid  = threadIdx.x;
    const int lane = tid & 31;
    const int w    = tid >> 5;
    const int waveRow = w >> 2;        // 0..1 -> 32 rows each
    const int waveCol = w & 3;         // 0..3 -> 64 cols each
    const int rowBase = blockIdx.x * 64;
    const int im = blockIdx.z % MM;
    const int b  = blockIdx.z / MM;

    const float*  Eblk  = E  + ((size_t)b * NN + (size_t)im * NLOC) * NN + (size_t)im * NLOC;
    const size_t  nodeBase = (size_t)b * NN + (size_t)im * NLOC;
    const bf16_t* XTblk = xT + ((size_t)b * MM + im) * CC * NLOC;   // [c][node]

    __shared__ __align__(16) bf16_t ldsE[64][32];
    __shared__ __align__(16) bf16_t ldsXT[256][32];
    __shared__ float rinv[64];    // becomes 1/max(L1-diag, eps)
    __shared__ float rdiag[64];

    if (tid < 64) rinv[tid] = 0.0f;

    v8f acc[2][4];
    #pragma unroll
    for (int mi = 0; mi < 2; ++mi)
        #pragma unroll
        for (int ni = 0; ni < 4; ++ni)
            #pragma unroll
            for (int r = 0; r < 8; ++r) acc[mi][ni][r] = 0.0f;

    const int lrow = lane & 15;
    const int lhi  = lane >> 4;

    // per-thread E rows (fixed across the K loop): rows tid>>3 and 32+(tid>>3)
    float psum[2]  = {0.0f, 0.0f};
    const int erow0 = tid >> 3;          // c=0 row
    const int ek4   = tid & 7;           // float4 slot within the 32-wide K tile

    for (int kk = 0; kk < NLOC; kk += 32) {
        // Stage E tile (64 x 32) via float4 loads; fold |.| row sums (incl. diag)
        #pragma unroll
        for (int c = 0; c < 2; ++c) {
            int row = erow0 + c * 32;
            const float4 ev = *(const float4*)&Eblk[(size_t)(rowBase + row) * NN + kk + ek4 * 4];
            psum[c] += fabsf(ev.x) + fabsf(ev.y) + fabsf(ev.z) + fabsf(ev.w);
            v4bf bv;
            bv[0] = (bf16_t)ev.x; bv[1] = (bf16_t)ev.y;
            bv[2] = (bf16_t)ev.z; bv[3] = (bf16_t)ev.w;
            *(v4bf*)&ldsE[row][ek4 * 4] = bv;
        }
        // Stage X^T tile (256 cols x 32 k) via b128 loads/stores (pre-transposed)
        #pragma unroll
        for (int c = 0; c < 4; ++c) {
            int idx = c * 256 + tid;
            int col = idx >> 2, v8 = idx & 3;
            v8bf xv = *(const v8bf*)&XTblk[(size_t)col * NLOC + kk + v8 * 8];
            *(v8bf*)&ldsXT[col][v8 * 8] = xv;
        }
        __syncthreads();

        // A fragments: lane row = lane&15, K = lhi*8 + {0..7} and +16..+23
        v16bf afrag[2];
        #pragma unroll
        for (int mi = 0; mi < 2; ++mi) {
            int r = waveRow * 32 + mi * 16 + lrow;
            const v8bf* p0 = (const v8bf*)&ldsE[r][lhi * 8];
            const v8bf* p1 = (const v8bf*)&ldsE[r][lhi * 8 + 16];
            #pragma unroll
            for (int j = 0; j < 8; ++j) { afrag[mi][j] = (*p0)[j]; afrag[mi][8 + j] = (*p1)[j]; }
        }
        // B fragments: lane col = lane&15, K = lhi*8 + {0..7} and +16..+23
        #pragma unroll
        for (int ni = 0; ni < 4; ++ni) {
            int col = waveCol * 64 + ni * 16 + lrow;
            const v8bf* q0 = (const v8bf*)&ldsXT[col][lhi * 8];
            const v8bf* q1 = (const v8bf*)&ldsXT[col][lhi * 8 + 16];
            v16bf bfrag;
            #pragma unroll
            for (int j = 0; j < 8; ++j) { bfrag[j] = (*q0)[j]; bfrag[8 + j] = (*q1)[j]; }
            #pragma unroll
            for (int mi = 0; mi < 2; ++mi)
                acc[mi][ni] = __builtin_amdgcn_wmma_f32_16x16x32_bf16(
                    false, afrag[mi], false, bfrag, (short)0, acc[mi][ni], false, false);
        }
        __syncthreads();
    }

    // Fold per-thread row partials; then one thread per row grabs the diagonal
    // directly and finalizes inv = 1/max(L1 - |diag|, eps).
    #pragma unroll
    for (int c = 0; c < 2; ++c)
        atomicAdd(&rinv[erow0 + c * 32], psum[c]);
    __syncthreads();
    if (tid < 64) {
        int row = rowBase + tid;
        float dgv = Eblk[(size_t)row * NN + row];
        rdiag[tid] = dgv;
        rinv[tid]  = 1.0f / fmaxf(rinv[tid] - fabsf(dgv), 1e-12f);
    }
    __syncthreads();

    // Epilogue: remove diagonal contribution, apply L1 scale, store bf16.
    #pragma unroll
    for (int mi = 0; mi < 2; ++mi) {
        #pragma unroll
        for (int ni = 0; ni < 4; ++ni) {
            #pragma unroll
            for (int r = 0; r < 8; ++r) {
                int lrl = waveRow * 32 + mi * 16 + lhi * 8 + r;   // M = r + 8*lhi
                int lc  = waveCol * 64 + ni * 16 + lrow;          // N = lane&15
                size_t node = nodeBase + rowBase + lrl;
                float v = (acc[mi][ni][r] - rdiag[lrl] * xf[node * CC + lc]) * rinv[lrl];
                P[node * CC + lc] = (bf16_t)v;
            }
        }
    }
}

// ---------------------------------------------------------------------------
// Kernel C: Out = A @ W[i] + bias[i]  (A: n x C bf16 natural, WT: pre-transposed)
// Grid (n/128, C/128, B*M); 8 waves; wave tile 32x64 (2x4 wmma).
// ---------------------------------------------------------------------------
__global__ __launch_bounds__(256) void ncgemm_kernel(
    const bf16_t* __restrict__ A, const bf16_t* __restrict__ WT,
    const float* __restrict__ bias, float* __restrict__ Out)
{
    const int tid  = threadIdx.x;
    const int lane = tid & 31;
    const int w    = tid >> 5;
    const int waveRow = w >> 1;
    const int waveCol = w & 1;
    const int rowBase = blockIdx.x * 128;
    const int colBase = blockIdx.y * 128;
    const int im = blockIdx.z % MM;
    const int b  = blockIdx.z / MM;

    const size_t  nodeBase = (size_t)b * NN + (size_t)im * NLOC;
    const bf16_t* Ablk  = A  + nodeBase * CC;
    const bf16_t* WTblk = WT + (size_t)im * CC * CC;    // [c2][c1]

    __shared__ __align__(16) bf16_t ldsA[128][32];
    __shared__ __align__(16) bf16_t ldsWT[128][32];

    v8f acc[2][4];
    #pragma unroll
    for (int mi = 0; mi < 2; ++mi)
        #pragma unroll
        for (int ni = 0; ni < 4; ++ni)
            #pragma unroll
            for (int r = 0; r < 8; ++r) acc[mi][ni][r] = 0.0f;

    const int lrow = lane & 15;
    const int lhi  = lane >> 4;

    for (int kk = 0; kk < CC; kk += 32) {
        #pragma unroll
        for (int c = 0; c < 2; ++c) {
            int idx = c * 256 + tid;
            int row = idx >> 2, v8 = idx & 3;
            v8bf av = *(const v8bf*)&Ablk[(size_t)(rowBase + row) * CC + kk + v8 * 8];
            *(v8bf*)&ldsA[row][v8 * 8] = av;
        }
        #pragma unroll
        for (int c = 0; c < 2; ++c) {
            int idx = c * 256 + tid;
            int col = idx >> 2, v8 = idx & 3;
            v8bf wv = *(const v8bf*)&WTblk[(size_t)(colBase + col) * CC + kk + v8 * 8];
            *(v8bf*)&ldsWT[col][v8 * 8] = wv;
        }
        __syncthreads();

        v16bf afrag[2];
        #pragma unroll
        for (int mi = 0; mi < 2; ++mi) {
            int r = waveRow * 32 + mi * 16 + lrow;
            const v8bf* p0 = (const v8bf*)&ldsA[r][lhi * 8];
            const v8bf* p1 = (const v8bf*)&ldsA[r][lhi * 8 + 16];
            #pragma unroll
            for (int j = 0; j < 8; ++j) { afrag[mi][j] = (*p0)[j]; afrag[mi][8 + j] = (*p1)[j]; }
        }
        #pragma unroll
        for (int ni = 0; ni < 4; ++ni) {
            int col = waveCol * 64 + ni * 16 + lrow;
            const v8bf* q0 = (const v8bf*)&ldsWT[col][lhi * 8];
            const v8bf* q1 = (const v8bf*)&ldsWT[col][lhi * 8 + 16];
            v16bf bfrag;
            #pragma unroll
            for (int j = 0; j < 8; ++j) { bfrag[j] = (*q0)[j]; bfrag[8 + j] = (*q1)[j]; }
            #pragma unroll
            for (int mi = 0; mi < 2; ++mi)
                acc[mi][ni] = __builtin_amdgcn_wmma_f32_16x16x32_bf16(
                    false, afrag[mi], false, bfrag, (short)0, acc[mi][ni], false, false);
        }
        __syncthreads();
    }

    #pragma unroll
    for (int mi = 0; mi < 2; ++mi)
        #pragma unroll
        for (int ni = 0; ni < 4; ++ni)
            #pragma unroll
            for (int r = 0; r < 8; ++r) {
                int lrl = rowBase + waveRow * 32 + mi * 16 + lhi * 8 + r;
                int lc  = colBase + waveCol * 64 + ni * 16 + lrow;
                size_t node = nodeBase + lrl;
                Out[node * CC + lc] = acc[mi][ni][r] + bias[(size_t)im * CC + lc];
            }
}

// ---------------------------------------------------------------------------
// Kernel D: gate (sigmoid of folded W_beta dot), LayerNorm, ReLU.
// One block per (b, node); 256 threads = C channels.
// ---------------------------------------------------------------------------
__global__ __launch_bounds__(256) void epilogue_kernel(
    const float* __restrict__ feat, const float* __restrict__ aggr,
    const float* __restrict__ Wbeta, const float* __restrict__ bbeta,
    const float* __restrict__ gamma, const float* __restrict__ lnb,
    float* __restrict__ out)
{
    const int nodeG = blockIdx.x;           // b*N + node
    const int node  = nodeG % NN;
    const int im    = node / NLOC;
    const int c     = threadIdx.x;
    const size_t base = (size_t)nodeG * CC;

    float f = feat[base + c];
    float a = aggr[base + c];
    const float* wb = Wbeta + (size_t)im * 3 * CC;
    // cat@Wb = dot(feat, wb1+wb3) + dot(aggr, wb2-wb3)
    float part = f * (wb[c] + wb[2 * CC + c]) + a * (wb[CC + c] - wb[2 * CC + c]);

    __shared__ float red[256];
    red[c] = part; __syncthreads();
    for (int off = 128; off > 0; off >>= 1) {
        if (c < off) red[c] += red[c + off];
        __syncthreads();
    }
    float s = red[0];
    float beta = 1.0f / (1.0f + expf(-(s + bbeta[im])));
    float h = f * beta + a * (1.0f - beta);

    __syncthreads();
    red[c] = h; __syncthreads();
    for (int off = 128; off > 0; off >>= 1) {
        if (c < off) red[c] += red[c + off];
        __syncthreads();
    }
    float mu = red[0] * (1.0f / CC);

    __syncthreads();
    float d = h - mu;
    red[c] = d * d; __syncthreads();
    for (int off = 128; off > 0; off >>= 1) {
        if (c < off) red[c] += red[c + off];
        __syncthreads();
    }
    float var = red[0] * (1.0f / CC);

    float y = d * rsqrtf(var + 1e-5f) * gamma[(size_t)im * CC + c] + lnb[(size_t)im * CC + c];
    out[base + c] = fmaxf(y, 0.0f);
}

// ---------------------------------------------------------------------------
extern "C" void kernel_launch(void* const* d_in, const int* in_sizes, int n_in,
                              void* d_out, int out_size, void* d_ws, size_t ws_size,
                              hipStream_t stream)
{
    const float* E      = (const float*)d_in[0];
    const float* xf     = (const float*)d_in[1];
    // d_in[2] modal_id: layout is fixed block-diagonal, unused.
    const float* Wf     = (const float*)d_in[3];
    const float* bfeat  = (const float*)d_in[4];
    const float* Wr     = (const float*)d_in[5];
    const float* braw   = (const float*)d_in[6];
    const float* Wbeta  = (const float*)d_in[7];
    const float* bbeta  = (const float*)d_in[8];
    const float* gamma  = (const float*)d_in[9];
    const float* lnb    = (const float*)d_in[10];
    float* out = (float*)d_out;

    char* ws = (char*)d_ws;
    size_t off = 0;
    auto alloc = [&](size_t bytes) { char* p = ws + off; off += (bytes + 255) & ~(size_t)255; return p; };

    bf16_t* xb    = (bf16_t*)alloc((size_t)BB * NN * CC * sizeof(bf16_t));   // x, natural
    bf16_t* xT    = (bf16_t*)alloc((size_t)BB * NN * CC * sizeof(bf16_t));   // x, transposed
    bf16_t* wfbT  = (bf16_t*)alloc((size_t)MM * CC * CC * sizeof(bf16_t));
    bf16_t* wrbT  = (bf16_t*)alloc((size_t)MM * CC * CC * sizeof(bf16_t));
    bf16_t* P     = (bf16_t*)alloc((size_t)BB * NN * CC * sizeof(bf16_t));
    float*  featw = (float*) alloc((size_t)BB * NN * CC * sizeof(float));
    float*  aggrw = (float*) alloc((size_t)BB * NN * CC * sizeof(float));

    // Conversions / transposes
    {
        int nx = BB * NN * CC;                 // 6,291,456
        f2bf_kernel<<<(nx + 255) / 256, 256, 0, stream>>>(xf, xb, nx);
        xT_kernel<<<(nx + 255) / 256, 256, 0, stream>>>(xf, xT);
        int nw = MM * CC * CC;                 // 196,608
        wT_kernel<<<(nw + 255) / 256, 256, 0, stream>>>(Wf, wfbT);
        wT_kernel<<<(nw + 255) / 256, 256, 0, stream>>>(Wr, wrbT);
    }

    // P = L1-normalized(E, diag=0) @ X : fused row-norm + bf16 WMMA, one E pass
    {
        dim3 grid(NLOC / 64, 1, BB * MM);
        eggemm_kernel<<<grid, 256, 0, stream>>>(E, xT, xf, P);
    }

    // feat = X @ W_feat + b_feat ; aggr = P @ W_raw + b_raw
    {
        dim3 grid(NLOC / 128, CC / 128, BB * MM);
        ncgemm_kernel<<<grid, 256, 0, stream>>>(xb, wfbT, bfeat, featw);
        ncgemm_kernel<<<grid, 256, 0, stream>>>(P,  wrbT, braw,  aggrw);
    }

    // Gate + LayerNorm + ReLU
    epilogue_kernel<<<BB * NN, 256, 0, stream>>>(featw, aggrw, Wbeta, bbeta, gamma, lnb, out);
}
